// Model_41781441856316
// MI455X (gfx1250) — matrix operand order repaired and checked by reading, not compiled, not run on hardware
//
#include <hip/hip_runtime.h>
#include <hip/hip_fp16.h>
#include <math.h>

typedef __attribute__((ext_vector_type(16))) _Float16 v16h;
typedef __attribute__((ext_vector_type(8)))  _Float16 v8h;
typedef __attribute__((ext_vector_type(8)))  float    v8f;

#define N_ATOMS 4096
#define DEG     48
#define DPET    128
#define NEDGE   (N_ATOMS*DEG)
#define NSTR    64

// LDS row strides, padded so stride(bytes)/4 mod 64 == 4 -> 16 consecutive rows
// hit 16 distinct bank groups (64-bank LDS), while keeping 16B alignment for b128 loads.
#define XS 132   // x32 f32 stride (528B = 132 dw; 132%64=4)
#define NS 136   // xn  f16 stride (272B = 68 dw)
#define QS 392   // qkv f16 stride (784B = 196 dw)
#define HS 520   // mlp hidden f16 stride (1040B = 260 dw)
#define WS 72    // softmax-weight f16 stride (144B = 36 dw)
#define AOS 136  // attnout f16 stride
#define CS 264   // contraction A f16 stride (528B = 132 dw)

// ---------------- WMMA lane-layout helpers (gfx1250, wave32) ----------------
// A 16x32 f16: lane = hf*16+m holds row m, K = hf*16 + j
static __device__ __forceinline__ v16h load_a16(const _Float16* base, int lda) {
  const int lane = threadIdx.x & 31;
  const int m = lane & 15, hf = lane >> 4;
  const _Float16* p = base + m * lda + hf * 16;
  v8h lo = *(const v8h*)(p);
  v8h hi = *(const v8h*)(p + 8);
  return __builtin_shufflevector(lo, hi, 0,1,2,3,4,5,6,7,8,9,10,11,12,13,14,15);
}
// B 32x16 f16 from pre-packed tile: lane-contiguous 16 halves (single 32B load)
static __device__ __forceinline__ v16h load_b_packed(const _Float16* tileBase) {
  const int lane = threadIdx.x & 31;
  return *(const v16h*)(tileBase + lane * 16);
}
// B 32x16 where the source is row-major S[n][k] (i.e. B = S^T): contiguous in k
static __device__ __forceinline__ v16h load_b_ct(const _Float16* base, int ldb) {
  const int lane = threadIdx.x & 31;
  const int n = lane & 15, hf = lane >> 4;
  const _Float16* p = base + n * ldb + hf * 16;
  v8h lo = *(const v8h*)(p);
  v8h hi = *(const v8h*)(p + 8);
  return __builtin_shufflevector(lo, hi, 0,1,2,3,4,5,6,7,8,9,10,11,12,13,14,15);
}
// B 32x16 row-major B[k][n] (strided gather)
static __device__ __forceinline__ v16h load_b_rm(const _Float16* base, int ldb) {
  const int lane = threadIdx.x & 31;
  const int n = lane & 15, hf = lane >> 4;
  v16h b;
#pragma unroll
  for (int j = 0; j < 16; ++j) b[j] = base[(hf * 16 + j) * ldb + n];
  return b;
}
// C/D f32 16x16: VGPR r: lanes0-15 -> M=r, lanes16-31 -> M=8+r; N = lane&15
static __device__ __forceinline__ v8f load_c_f32(const float* base, int ldc) {
  const int lane = threadIdx.x & 31;
  const int n = lane & 15, mh = (lane >> 4) * 8;
  v8f c;
#pragma unroll
  for (int r = 0; r < 8; ++r) c[r] = base[(mh + r) * ldc + n];
  return c;
}
static __device__ __forceinline__ void store_c_f16(v8f c, _Float16* base, int ldc) {
  const int lane = threadIdx.x & 31;
  const int n = lane & 15, mh = (lane >> 4) * 8;
#pragma unroll
  for (int r = 0; r < 8; ++r) base[(mh + r) * ldc + n] = (_Float16)c[r];
}
static __device__ __forceinline__ v8f wmma16(v16h a, v16h b, v8f c) {
  return __builtin_amdgcn_wmma_f32_16x16x32_f16(false, a, false, b, (short)0, c, false, false);
}

// ---------------- small utility kernels ----------------
__global__ void pet_zero_i32(int* p, int n) {
  int i = blockIdx.x * 256 + threadIdx.x;
  if (i < n) p[i] = 0;
}

// Pack a K x N f32 weight into WMMA-B tile layout (f16): per (32x16) tile, 32 "lanes" x 16 halves.
__global__ void pet_pack_b(const float* __restrict__ W, _Float16* __restrict__ out, int K, int N) {
  int idx = blockIdx.x * 256 + threadIdx.x;
  if (idx >= K * N) return;
  int k = idx / N, n = idx % N;
  int kt = k >> 5, kr = k & 31, hf = kr >> 4, j = kr & 15;
  int nt = n >> 4, nl = n & 15, lane = hf * 16 + nl;
  out[(((size_t)(kt * (N >> 4) + nt)) * 32 + lane) * 16 + j] = (_Float16)W[idx];
}

// Fold encoder: cartc = cart_w @ comp_w[0:128], cenc = center_emb @ comp_w[128:256], neic = neighbor_emb @ comp_w[256:384]
__global__ void pet_precomp(const float* __restrict__ cart_w, const float* __restrict__ cemb,
                            const float* __restrict__ nemb, const float* __restrict__ comp_w,
                            float* __restrict__ cartc, float* __restrict__ cenc, float* __restrict__ neic) {
  int t = blockIdx.x * 256 + threadIdx.x;
  if (t < 3 * 128) {
    int r = t / 128, d = t % 128; float s = 0.f;
    for (int j = 0; j < 128; ++j) s += cart_w[r * 128 + j] * comp_w[j * 128 + d];
    cartc[t] = s;
  } else if (t < 3 * 128 + 4 * 128) {
    int i = t - 3 * 128; int r = i / 128, d = i % 128; float s = 0.f;
    for (int j = 0; j < 128; ++j) s += cemb[r * 128 + j] * comp_w[(128 + j) * 128 + d];
    cenc[i] = s;
  } else if (t < 3 * 128 + 8 * 128) {
    int i = t - 3 * 128 - 4 * 128; int r = i / 128, d = i % 128; float s = 0.f;
    for (int j = 0; j < 128; ++j) s += nemb[r * 128 + j] * comp_w[(256 + j) * 128 + d];
    neic[i] = s;
  }
}

// bucket-fill: slot per center (order arbitrary; sorted next pass -> deterministic)
__global__ void pet_build1(const int* __restrict__ centers, int* __restrict__ cnt,
                           int* __restrict__ tmp, int E) {
  int e = blockIdx.x * 256 + threadIdx.x;
  if (e >= E) return;
  int c = centers[e];
  int p = atomicAdd(&cnt[c], 1);
  tmp[c * DEG + p] = e;
}

// per-atom: sort 48 edge ids ascending (== rank by original index, matches stable argsort)
__global__ void pet_build2(const int* __restrict__ tmp, const int* __restrict__ neighbors,
                           const int* __restrict__ species, int* __restrict__ nefE,
                           int* __restrict__ nbr, int* __restrict__ elem, int N) {
  int a = blockIdx.x * 256 + threadIdx.x;
  if (a >= N) return;
  int v[DEG];
  for (int k = 0; k < DEG; ++k) v[k] = tmp[a * DEG + k];
  for (int i = 1; i < DEG; ++i) {
    int key = v[i], j = i - 1;
    while (j >= 0 && v[j] > key) { v[j + 1] = v[j]; --j; }
    v[j + 1] = key;
  }
  for (int k = 0; k < DEG; ++k) {
    nefE[a * DEG + k] = v[k];
    nbr[a * DEG + k] = neighbors[v[k]];
  }
  int sp = species[a];
  elem[a] = (sp == 1) ? 0 : (sp == 6) ? 1 : (sp == 7) ? 2 : 3;
}

__global__ void pet_build3(const float* __restrict__ ev, const int* __restrict__ nefE,
                           float* __restrict__ rmo, int total) {
  int i = blockIdx.x * 256 + threadIdx.x;
  if (i >= total) return;
  int e = nefE[i];
  float x = ev[e * 3], y = ev[e * 3 + 1], z = ev[e * 3 + 2];
  float r = sqrtf(x * x + y * y + z * z);
  rmo[i] = (r < 3.0f) ? 1.0f : 0.5f * (cosf(3.14159265358979323846f * (r - 3.0f) * 0.5f) + 1.0f);
}

// reverse-edge map: rev[a*48+k] = n*48+k2 with nbr[n][k2]==a
__global__ void pet_build4(const int* __restrict__ nbr, int* __restrict__ rev, int total) {
  int i = blockIdx.x * 256 + threadIdx.x;
  if (i >= total) return;
  int a = i / DEG;
  int n = nbr[i];
  int k2 = 0;
  for (int j = 0; j < DEG; ++j)
    if (nbr[n * DEG + j] == a) k2 = j;
  rev[i] = n * DEG + k2;
}

// encoder: F[a][k][d] = ev . cartc + cenc[elem[a]] + neic[elem[nbr]]
__global__ void pet_encode(const float* __restrict__ ev, const int* __restrict__ nefE,
                           const int* __restrict__ nbr, const int* __restrict__ elem,
                           const float* __restrict__ cartc, const float* __restrict__ cenc,
                           const float* __restrict__ neic, _Float16* __restrict__ F) {
  long idx = (long)blockIdx.x * 256 + threadIdx.x;
  if (idx >= (long)NEDGE * DPET) return;
  int d = (int)(idx & 127);
  long ak = idx >> 7;
  int a = (int)(ak / DEG);
  int e = nefE[ak];
  float x = ev[e * 3], y = ev[e * 3 + 1], z = ev[e * 3 + 2];
  int ci = elem[a], ni = elem[nbr[ak]];
  float v = x * cartc[d] + y * cartc[128 + d] + z * cartc[256 + d] + cenc[ci * 128 + d] + neic[ni * 128 + d];
  F[idx] = (_Float16)v;
}

// ---------------- fused transformer (2 layers), one workgroup per atom ----------------
struct LayerP {
  const _Float16 *qkv, *ow, *w1, *w2;     // packed f16 B-layout weights
  const float *ln1s, *ln1b, *ln2s, *ln2b, *b1, *b2;
};

static __device__ __forceinline__ void layer_norm48(const float* __restrict__ x32,
                                                    _Float16* __restrict__ xn,
                                                    const float* __restrict__ s,
                                                    const float* __restrict__ b, int tid) {
  for (int row = tid; row < 48; row += 128) {
    const float4* xr4 = (const float4*)(x32 + row * XS);
    float m = 0.f;
#pragma unroll
    for (int i = 0; i < 32; ++i) { float4 v = xr4[i]; m += v.x + v.y + v.z + v.w; }
    m *= (1.f / 128.f);
    float var = 0.f;
#pragma unroll
    for (int i = 0; i < 32; ++i) {
      float4 v = xr4[i];
      float a0 = v.x - m, a1 = v.y - m, a2 = v.z - m, a3 = v.w - m;
      var += a0 * a0 + a1 * a1 + a2 * a2 + a3 * a3;
    }
    float inv = rsqrtf(var * (1.f / 128.f) + 1e-5f);
    const float* xr = x32 + row * XS;
    _Float16* o = xn + row * NS;
    for (int d = 0; d < 128; ++d) o[d] = (_Float16)((xr[d] - m) * inv * s[d] + b[d]);
  }
}

__global__ __launch_bounds__(128) void pet_transformer(_Float16* __restrict__ X,
                                                       const float* __restrict__ rm,
                                                       LayerP L0, LayerP L1) {
  __shared__ __align__(16) float    x32[48 * XS];       // residual stream (f32), padded stride
  __shared__ __align__(16) _Float16 xn[48 * NS];        // normalized activations (f16)
  __shared__ __align__(16) _Float16 r1[64 * QS];        // qkv (64 rows padded) / mlp hidden (48 x HS)
  __shared__ __align__(16) float    scores[4 * 48 * 48];// attention scores f32 / attnout f16 alias
  __shared__ __align__(16) _Float16 wsm[4 * 48 * WS];   // softmax*mask weights, K padded to 64

  const int a = blockIdx.x;
  const int tid = threadIdx.x;
  const int wave = tid >> 5;
  _Float16* Xa = X + (size_t)a * 48 * 128;
  const float* rma = rm + a * 48;

  for (int i = tid; i < 48 * 128; i += 128) {
    int row = i >> 7, d = i & 127;
    x32[row * XS + d] = (float)Xa[i];
  }

  for (int li = 0; li < 2; ++li) {
    LayerP P = li ? L1 : L0;
    __syncthreads();
    // zero qkv pad rows 48..63 (r1 is clobbered by mlp hidden each layer)
    for (int i = tid; i < 16 * QS; i += 128) r1[48 * QS + i] = (_Float16)0.f;
    layer_norm48(x32, xn, P.ln1s, P.ln1b, tid);
    __syncthreads();
    // --- QKV GEMM: (48x128)@(128x384), A tiles hoisted to VGPRs ---
    for (int mt = 0; mt < 3; ++mt) {
      v16h a0 = load_a16(xn + mt * 16 * NS + 0 * 32, NS);
      v16h a1 = load_a16(xn + mt * 16 * NS + 1 * 32, NS);
      v16h a2 = load_a16(xn + mt * 16 * NS + 2 * 32, NS);
      v16h a3 = load_a16(xn + mt * 16 * NS + 3 * 32, NS);
      for (int nt = wave; nt < 24; nt += 4) {
        v8f c = {};
        c = wmma16(a0, load_b_packed(P.qkv + (size_t)(0 * 24 + nt) * 512), c);
        c = wmma16(a1, load_b_packed(P.qkv + (size_t)(1 * 24 + nt) * 512), c);
        c = wmma16(a2, load_b_packed(P.qkv + (size_t)(2 * 24 + nt) * 512), c);
        c = wmma16(a3, load_b_packed(P.qkv + (size_t)(3 * 24 + nt) * 512), c);
        store_c_f16(c, r1 + mt * 16 * QS + nt * 16, QS);
      }
    }
    __syncthreads();
    // --- scores = q k^T / sqrt(32), per head ---
    for (int t = wave; t < 12; t += 4) {
      int h = t / 3, mt = t % 3;
      v16h av = load_a16(r1 + mt * 16 * QS + h * 32, QS);
      const int lane = tid & 31, n = lane & 15, mh = (lane >> 4) * 8;
      for (int nt = 0; nt < 3; ++nt) {
        v16h bv = load_b_ct(r1 + nt * 16 * QS + 128 + h * 32, QS);
        v8f c = {};
        c = wmma16(av, bv, c);
#pragma unroll
        for (int r = 0; r < 8; ++r)
          scores[(h * 48 + mt * 16 + mh + r) * 48 + nt * 16 + n] = c[r] * 0.17677669529663687f;
      }
    }
    __syncthreads();
    // --- softmax then * radial mask; write K-padded f16 weights ---
    for (int row = tid; row < 4 * 48; row += 128) {
      int h = row / 48, q = row % 48;
      float* sr = scores + (h * 48 + q) * 48;
      float mx = sr[0];
      for (int k = 1; k < 48; ++k) mx = fmaxf(mx, sr[k]);
      float s = 0.f;
      for (int k = 0; k < 48; ++k) { float e = expf(sr[k] - mx); sr[k] = e; s += e; }
      float inv = 1.f / s;
      _Float16* wr = wsm + (h * 48 + q) * WS;
      for (int k = 0; k < 48; ++k) wr[k] = (_Float16)(sr[k] * inv * rma[k]);
      for (int k = 48; k < 64; ++k) wr[k] = (_Float16)0.f;
    }
    __syncthreads();
    // --- attn_out = w @ v  (K padded 48->64; v pad rows are zero) ---
    _Float16* attnout = (_Float16*)scores;  // scores dead, reuse region
    for (int t = wave; t < 12; t += 4) {
      int h = t / 3, mt = t % 3;
      v16h a0 = load_a16(wsm + (h * 48 + mt * 16) * WS + 0, WS);
      v16h a1 = load_a16(wsm + (h * 48 + mt * 16) * WS + 32, WS);
#pragma unroll
      for (int vt = 0; vt < 2; ++vt) {
        v8f c = {};
        c = wmma16(a0, load_b_rm(r1 + 256 + h * 32 + vt * 16, QS), c);
        c = wmma16(a1, load_b_rm(r1 + 32 * QS + 256 + h * 32 + vt * 16, QS), c);
        store_c_f16(c, attnout + mt * 16 * AOS + h * 32 + vt * 16, AOS);
      }
    }
    __syncthreads();
    // --- x += attn_out @ out_w  (residual as C-init) ---
    for (int mt = 0; mt < 3; ++mt) {
      v16h a0 = load_a16(attnout + mt * 16 * AOS + 0 * 32, AOS);
      v16h a1 = load_a16(attnout + mt * 16 * AOS + 1 * 32, AOS);
      v16h a2 = load_a16(attnout + mt * 16 * AOS + 2 * 32, AOS);
      v16h a3 = load_a16(attnout + mt * 16 * AOS + 3 * 32, AOS);
      for (int nt = wave; nt < 8; nt += 4) {
        v8f c = load_c_f32(x32 + mt * 16 * XS + nt * 16, XS);
        c = wmma16(a0, load_b_packed(P.ow + (size_t)(0 * 8 + nt) * 512), c);
        c = wmma16(a1, load_b_packed(P.ow + (size_t)(1 * 8 + nt) * 512), c);
        c = wmma16(a2, load_b_packed(P.ow + (size_t)(2 * 8 + nt) * 512), c);
        c = wmma16(a3, load_b_packed(P.ow + (size_t)(3 * 8 + nt) * 512), c);
        const int lane = tid & 31, n = lane & 15, mh = (lane >> 4) * 8;
#pragma unroll
        for (int r = 0; r < 8; ++r) x32[(mt * 16 + mh + r) * XS + nt * 16 + n] = c[r];
      }
    }
    __syncthreads();
    layer_norm48(x32, xn, P.ln2s, P.ln2b, tid);
    __syncthreads();
    // --- h = gelu(xn @ w1 + b1)  (exact erf GELU fused into epilogue) ---
    _Float16* hbuf = r1;  // overwrites qkv (dead)
    for (int mt = 0; mt < 3; ++mt) {
      v16h a0 = load_a16(xn + mt * 16 * NS + 0 * 32, NS);
      v16h a1 = load_a16(xn + mt * 16 * NS + 1 * 32, NS);
      v16h a2 = load_a16(xn + mt * 16 * NS + 2 * 32, NS);
      v16h a3 = load_a16(xn + mt * 16 * NS + 3 * 32, NS);
      for (int nt = wave; nt < 32; nt += 4) {
        v8f c = {};
        c = wmma16(a0, load_b_packed(P.w1 + (size_t)(0 * 32 + nt) * 512), c);
        c = wmma16(a1, load_b_packed(P.w1 + (size_t)(1 * 32 + nt) * 512), c);
        c = wmma16(a2, load_b_packed(P.w1 + (size_t)(2 * 32 + nt) * 512), c);
        c = wmma16(a3, load_b_packed(P.w1 + (size_t)(3 * 32 + nt) * 512), c);
        const int lane = tid & 31, n = lane & 15, mh = (lane >> 4) * 8;
#pragma unroll
        for (int r = 0; r < 8; ++r) {
          float v = c[r] + P.b1[nt * 16 + n];
          float g = 0.5f * v * (1.0f + erff(v * 0.70710678118654752f));
          hbuf[(mt * 16 + mh + r) * HS + nt * 16 + n] = (_Float16)g;
        }
      }
    }
    __syncthreads();
    // --- x += h @ w2 + b2 (two column tiles per wave, A shared across both) ---
    for (int mt = 0; mt < 3; ++mt) {
      const int nt0 = wave, nt1 = wave + 4;
      v8f c0 = load_c_f32(x32 + mt * 16 * XS + nt0 * 16, XS);
      v8f c1 = load_c_f32(x32 + mt * 16 * XS + nt1 * 16, XS);
      for (int kt = 0; kt < 16; ++kt) {
        v16h av = load_a16(hbuf + mt * 16 * HS + kt * 32, HS);
        c0 = wmma16(av, load_b_packed(P.w2 + (size_t)(kt * 8 + nt0) * 512), c0);
        c1 = wmma16(av, load_b_packed(P.w2 + (size_t)(kt * 8 + nt1) * 512), c1);
      }
      const int lane = tid & 31, n = lane & 15, mh = (lane >> 4) * 8;
#pragma unroll
      for (int r = 0; r < 8; ++r) {
        x32[(mt * 16 + mh + r) * XS + nt0 * 16 + n] = c0[r] + P.b2[nt0 * 16 + n];
        x32[(mt * 16 + mh + r) * XS + nt1 * 16 + n] = c1[r] + P.b2[nt1 * 16 + n];
      }
    }
    __syncthreads();
  }
  for (int i = tid; i < 48 * 128; i += 128) {
    int row = i >> 7, d = i & 127;
    Xa[i] = (_Float16)x32[row * XS + d];
  }
}

// ---------------- contraction: G[a][k] = concat(F[a][k], F[rev]) @ Wc (256->128) ----------------
__global__ __launch_bounds__(128) void pet_contraction(const _Float16* __restrict__ F,
                                                       const int* __restrict__ rev,
                                                       const _Float16* __restrict__ Wc,
                                                       _Float16* __restrict__ G) {
  __shared__ __align__(16) _Float16 A[48 * CS];
  const int a = blockIdx.x, tid = threadIdx.x, wave = tid >> 5;
  for (int i = tid; i < 48 * 128; i += 128) {
    int k = i >> 7, d = i & 127;
    A[k * CS + d]       = F[((size_t)a * DEG + k) * 128 + d];
    A[k * CS + 128 + d] = F[(size_t)rev[a * DEG + k] * 128 + d];
  }
  __syncthreads();
  for (int mt = 0; mt < 3; ++mt) {
    const int nt0 = wave, nt1 = wave + 4;
    v8f c0 = {}, c1 = {};
    for (int kt = 0; kt < 8; ++kt) {
      v16h av = load_a16(A + mt * 16 * CS + kt * 32, CS);
      c0 = wmma16(av, load_b_packed(Wc + (size_t)(kt * 8 + nt0) * 512), c0);
      c1 = wmma16(av, load_b_packed(Wc + (size_t)(kt * 8 + nt1) * 512), c1);
    }
    const int lane = tid & 31, n = lane & 15, mh = (lane >> 4) * 8;
    _Float16* gp = G + ((size_t)a * DEG + mt * 16) * 128;
#pragma unroll
    for (int r = 0; r < 8; ++r) {
      gp[(mh + r) * 128 + nt0 * 16 + n] = (_Float16)c0[r];
      gp[(mh + r) * 128 + nt1 * 16 + n] = (_Float16)c1[r];
    }
  }
}

// ---------------- readout + structure/composition reduction ----------------
__global__ __launch_bounds__(256) void pet_readout(const _Float16* __restrict__ F,
                                                   const _Float16* __restrict__ G,
                                                   const float* __restrict__ rm,
                                                   const float* __restrict__ rw,
                                                   const int* __restrict__ elem,
                                                   const float* __restrict__ compw,
                                                   float* __restrict__ out) {
  __shared__ float red[256];
  const int s = blockIdx.x, tid = threadIdx.x;
  float acc = 0.f;
  for (int a = s * 64; a < s * 64 + 64; ++a) {
    const _Float16* Fa = F + (size_t)a * 48 * 128;
    const _Float16* Ga = G + (size_t)a * 48 * 128;
    const float* rma = rm + a * 48;
    for (int i = tid; i < 48 * 128; i += 256) {
      int k = i >> 7, d = i & 127;
      acc += ((float)Fa[i] + (float)Ga[i]) * rw[d] * rma[k];
    }
  }
  red[tid] = acc;
  __syncthreads();
  for (int o = 128; o > 0; o >>= 1) {
    if (tid < o) red[tid] += red[tid + o];
    __syncthreads();
  }
  if (tid == 0) {
    float e = red[0];
    for (int a = s * 64; a < s * 64 + 64; ++a) e += compw[elem[a]];
    out[s] = e;
  }
}

// ---------------- host launcher ----------------
extern "C" void kernel_launch(void* const* d_in, const int* in_sizes, int n_in,
                              void* d_out, int out_size, void* d_ws, size_t ws_size,
                              hipStream_t stream) {
  (void)in_sizes; (void)n_in; (void)out_size; (void)ws_size;
  // Input order (setup_inputs insertion order, nested dicts/lists flattened in order):
  const float* ev        = (const float*)d_in[0];   // edge_vectors [E][3]
  const int*   centers   = (const int*)d_in[1];
  const int*   neighbors = (const int*)d_in[2];
  const int*   species   = (const int*)d_in[3];
  // d_in[4] = segment_indices (structures are contiguous blocks of 64 atoms)
  const float* cart_w    = (const float*)d_in[5];
  const float* cemb      = (const float*)d_in[6];
  const float* nemb      = (const float*)d_in[7];
  const float* comp_w    = (const float*)d_in[8];
  // transformer0 layers at 9..18, 19..28; contraction_w 29; transformer1 at 30..39, 40..49
  const float* contr_w   = (const float*)d_in[29];
  const float* readout_w = (const float*)d_in[50];
  const float* compw     = (const float*)d_in[51];
  float* out = (float*)d_out;

  char* ws = (char*)d_ws;
  size_t off = 0;
  auto take = [&](size_t bytes) -> void* {
    void* p = ws + off;
    off = (off + bytes + 255) & ~(size_t)255;
    return p;
  };
  _Float16* F    = (_Float16*)take((size_t)NEDGE * DPET * 2);
  _Float16* G    = (_Float16*)take((size_t)NEDGE * DPET * 2);
  _Float16* Wpk  = (_Float16*)take((size_t)(4 * 196608 + 32768) * 2);
  float*    cartc = (float*)take(3 * 128 * 4);
  float*    cenc  = (float*)take(4 * 128 * 4);
  float*    neic  = (float*)take(4 * 128 * 4);
  int*      nefE  = (int*)take((size_t)NEDGE * 4);
  int*      nbrN  = (int*)take((size_t)NEDGE * 4);
  int*      revI  = (int*)take((size_t)NEDGE * 4);
  float*    rmN   = (float*)take((size_t)NEDGE * 4);
  int*      cnt   = (int*)take((size_t)N_ATOMS * 4);
  int*      tmp   = (int*)take((size_t)NEDGE * 4);
  int*      elem  = (int*)take((size_t)N_ATOMS * 4);

  // --- pack weights to f16 WMMA-B tile layout ---
  auto pack = [&](const float* W, _Float16* o, int K, int N) {
    int total = K * N;
    pet_pack_b<<<(total + 255) / 256, 256, 0, stream>>>(W, o, K, N);
  };
  const int lbase[4] = {9, 19, 30, 40};
  LayerP LP[4];
  for (int l = 0; l < 4; ++l) {
    int b = lbase[l];
    _Float16* base = Wpk + (size_t)l * 196608;
    pack((const float*)d_in[b + 2], base + 0,      128, 384);  // qkv_w
    pack((const float*)d_in[b + 3], base + 49152,  128, 128);  // out_w
    pack((const float*)d_in[b + 6], base + 65536,  128, 512);  // mlp_w1
    pack((const float*)d_in[b + 8], base + 131072, 512, 128);  // mlp_w2
    LP[l].qkv = base;          LP[l].ow = base + 49152;
    LP[l].w1  = base + 65536;  LP[l].w2 = base + 131072;
    LP[l].ln1s = (const float*)d_in[b + 0]; LP[l].ln1b = (const float*)d_in[b + 1];
    LP[l].ln2s = (const float*)d_in[b + 4]; LP[l].ln2b = (const float*)d_in[b + 5];
    LP[l].b1   = (const float*)d_in[b + 7]; LP[l].b2   = (const float*)d_in[b + 9];
  }
  _Float16* Wcp = Wpk + (size_t)4 * 196608;
  pack(contr_w, Wcp, 256, 128);

  pet_precomp<<<6, 256, 0, stream>>>(cart_w, cemb, nemb, comp_w, cartc, cenc, neic);

  // --- neighbor-list structures ---
  pet_zero_i32<<<(N_ATOMS + 255) / 256, 256, 0, stream>>>(cnt, N_ATOMS);
  pet_build1<<<(NEDGE + 255) / 256, 256, 0, stream>>>(centers, cnt, tmp, NEDGE);
  pet_build2<<<(N_ATOMS + 255) / 256, 256, 0, stream>>>(tmp, neighbors, species, nefE, nbrN, elem, N_ATOMS);
  pet_build3<<<(NEDGE + 255) / 256, 256, 0, stream>>>(ev, nefE, rmN, NEDGE);
  pet_build4<<<(NEDGE + 255) / 256, 256, 0, stream>>>(nbrN, revI, NEDGE);

  // --- encoder ---
  {
    long total = (long)NEDGE * DPET;
    pet_encode<<<(unsigned)((total + 255) / 256), 256, 0, stream>>>(ev, nefE, nbrN, elem, cartc, cenc, neic, F);
  }

  // --- transformer 0 (in-place on F) ---
  pet_transformer<<<N_ATOMS, 128, 0, stream>>>(F, rmN, LP[0], LP[1]);
  // --- contraction F -> G ---
  pet_contraction<<<N_ATOMS, 128, 0, stream>>>(F, revI, Wcp, G);
  // --- transformer 1 (in-place on G) ---
  pet_transformer<<<N_ATOMS, 128, 0, stream>>>(G, rmN, LP[2], LP[3]);
  // --- readout: energies per structure ---
  pet_readout<<<NSTR, 256, 0, stream>>>(F, G, rmN, readout_w, elem, compw, out);
}
